// LSTMLosses_36833639531330
// MI455X (gfx1250) — matrix operand order repaired and checked by reading, compiled, not run on hardware
//
#include <hip/hip_runtime.h>
#include <hip/hip_bf16.h>
#include <stdint.h>

typedef _Float16 v16h __attribute__((ext_vector_type(16)));
typedef _Float16 v8h  __attribute__((ext_vector_type(8)));
typedef float    v8f  __attribute__((ext_vector_type(8)));

#define B_    16
#define L_    512
#define D_    256
#define NROWS (B_ * L_)      // 8192
#define NTILES (NROWS / 16)  // 512

union F16x16 { v16h v; v8h h[2]; };

// ---- gfx1250 async global->LDS staging --------------------------------------
// The clang builtin exists but its params are typed in the HIP `__device__` /
// `__shared__` language address spaces, which are not spellable from HIP
// source. Use inline asm (CDNA5 ISA §15.18, GLOBAL_LOAD_ASYNC_TO_LDS_B128):
//   global_load_async_to_lds_b128 vdst(lds byte addr), vaddr(64b), off
// Tracked by ASYNCcnt; completed with s_wait_asynccnt.
#define USE_ASYNC_ASM 1

__device__ __forceinline__ void stage16(const _Float16* __restrict__ src,
                                        _Float16* __restrict__ dst) {
#if USE_ASYNC_ASM
  uint32_t ldsoff = (uint32_t)(uintptr_t)dst;   // low 32 bits = LDS offset
  uint64_t gaddr  = (uint64_t)(uintptr_t)src;
  asm volatile("global_load_async_to_lds_b128 %0, %1, off"
               :: "v"(ldsoff), "v"(gaddr)
               : "memory");
#else
  *(v8h*)dst = *(const v8h*)src;                // global_load_b128 + ds_store_b128
#endif
}

__device__ __forceinline__ void async_stage_wait() {
#if USE_ASYNC_ASM
  asm volatile("s_wait_asynccnt 0x0" ::: "memory");
#endif
}

// ---------------------------------------------------------------------------
// Kernel 1: convert hidden -> f16 (layout preserved: row = 512 f16, fw|bw),
//           convert feats -> masked f16 (zero rows where l >= seq_len[b]).
// ---------------------------------------------------------------------------
__global__ void pack_kernel(const float* __restrict__ feats,
                            const float* __restrict__ hidden,
                            const long long* __restrict__ lens,
                            _Float16* __restrict__ fh,
                            _Float16* __restrict__ hh) {
  int idx = blockIdx.x * blockDim.x + threadIdx.x;   // 0 .. 4M-1
  if (idx < NROWS * 2 * D_) hh[idx] = (_Float16)hidden[idx];
  if (idx < NROWS * D_) {
    int row = idx / D_;          // b*L + l
    int b = row / L_;
    int l = row - b * L_;
    float v = (l < (int)lens[b]) ? feats[idx] : 0.0f;
    fh[idx] = (_Float16)v;
  }
}

// ---------------------------------------------------------------------------
// Kernel 2: den[dir][row] = sum_j exp( h_dir[row] . f[j] )
// Block: 256 threads = 8 waves; each wave owns one 16-row A tile in VGPRs and
// reuses it across all 512 N-tiles. B tiles (16 x 256 K f16 = 8KB) are staged
// into double-buffered LDS via async global->LDS and consumed by all 8 waves:
// one barrier per tile. wmma f32_16x16x32_f16, exp+rowsum epilogue.
// Grid: (NROWS/128, 2 directions).
// ---------------------------------------------------------------------------
__global__ void __launch_bounds__(256)
den_kernel(const _Float16* __restrict__ fh,
           const _Float16* __restrict__ hh,
           float* __restrict__ den) {
  __shared__ _Float16 ldsB[2][16][264];  // pad 8 f16 -> 528B stride, conflict-free

  const int dir  = blockIdx.y;
  const int wave = threadIdx.x >> 5;
  const int lane = threadIdx.x & 31;
  const int rowTile = blockIdx.x * 8 + wave;   // 8 row tiles per block
  const int row0 = rowTile * 16;
  const int m    = lane & 15;
  const int ksel = lane >> 4;                  // 0: low-K half, 1: high-K half

  // --- A fragments: 16x256 of hh for this tile/direction, kept in VGPRs ---
  // 16-bit A 16x32 layout: lanes 0-15 hold K {0..7, 16..23}; lanes 16-31 hold
  // K {8..15, 24..31} for row M=lane&15.
  const _Float16* aRow = hh + (size_t)(row0 + m) * (2 * D_) + dir * D_;
  F16x16 a[8];
#pragma unroll
  for (int ks = 0; ks < 8; ++ks) {
    const _Float16* p = aRow + ks * 32 + ksel * 8;
    a[ks].h[0] = *(const v8h*)(p);        // K + {0..7}   (or {8..15})
    a[ks].h[1] = *(const v8h*)(p + 16);   // K + {16..23} (or {24..31})
  }

  // Prologue: stage tile 0 into buffer 0 (512 chunks of 8 f16; 2 per thread).
  {
    int c = threadIdx.x;
#pragma unroll
    for (int r = 0; r < 2; ++r, c += 256) {
      int brow = c >> 5;
      int bcol = (c & 31) * 8;
      stage16(fh + (size_t)brow * D_ + bcol, &ldsB[0][brow][bcol]);
    }
  }
  async_stage_wait();
  __syncthreads();

  v8f dens = {0.f, 0.f, 0.f, 0.f, 0.f, 0.f, 0.f, 0.f};

  for (int nt = 0; nt < NTILES; ++nt) {
    const int p = nt & 1;
    // Prefetch next tile into the other buffer while computing this one.
    if (nt + 1 < NTILES) {
      int c = threadIdx.x;
#pragma unroll
      for (int r = 0; r < 2; ++r, c += 256) {
        int brow = c >> 5;
        int bcol = (c & 31) * 8;
        stage16(fh + (size_t)((nt + 1) * 16 + brow) * D_ + bcol,
                &ldsB[p ^ 1][brow][bcol]);
      }
    }

    v8f cacc = {0.f, 0.f, 0.f, 0.f, 0.f, 0.f, 0.f, 0.f};
#pragma unroll
    for (int ks = 0; ks < 8; ++ks) {
      // 16-bit B 32x16 layout: lanes 0-15 hold K {0..7, 16..23} for N=lane;
      // lanes 16-31 hold K {8..15, 24..31} for N=lane-16.
      F16x16 b;
      const _Float16* q = &ldsB[p][m][ks * 32 + ksel * 8];
      b.h[0] = *(const v8h*)(q);       // ds_load_b128
      b.h[1] = *(const v8h*)(q + 16);  // ds_load_b128
      cacc = __builtin_amdgcn_wmma_f32_16x16x32_f16(
          /*neg_a=*/false, a[ks].v, /*neg_b=*/false, b.v,
          /*c_mod=*/(short)0, cacc, /*reuse_a=*/false, /*reuse_b=*/false);
    }
    // Epilogue: exp + accumulate row sums (C layout: VGPR k = row k / k+8).
#pragma unroll
    for (int k = 0; k < 8; ++k) dens[k] += __expf(cacc[k]);

    async_stage_wait();   // our async writes to buf p^1 have landed in LDS
    __syncthreads();      // publish buf p^1; everyone done reading buf p
  }

  // Reduce over N=16 lanes within each half (xor offsets stay inside halves).
#pragma unroll
  for (int off = 1; off < 16; off <<= 1) {
#pragma unroll
    for (int k = 0; k < 8; ++k) dens[k] += __shfl_xor(dens[k], off, 32);
  }
  if ((lane & 15) == 0) {                // lane 0 -> rows 0..7, lane 16 -> 8..15
    int mbase = row0 + (lane >> 4) * 8;
#pragma unroll
    for (int k = 0; k < 8; ++k) den[dir * NROWS + mbase + k] = dens[k];
  }
}

// ---------------------------------------------------------------------------
// Kernel 3: per-(batch, dir) sequence score:
//   seq = sum_{l<len} exp(h[l] . f_shifted[l]) / den[l] ; partial = -log(seq)/len/B
// Grid (B, 2), 256 threads = 8 waves, one l per wave per iteration (f32 math).
// ---------------------------------------------------------------------------
__global__ void __launch_bounds__(256)
seq_kernel(const float* __restrict__ feats,
           const float* __restrict__ hidden,
           const long long* __restrict__ lens,
           const float* __restrict__ den,
           float* __restrict__ partial) {
  const int b    = blockIdx.x;
  const int dir  = blockIdx.y;
  const int wave = threadIdx.x >> 5;
  const int lane = threadIdx.x & 31;
  const int len  = (int)lens[b];
  __shared__ float sacc[8];

  float acc = 0.f;
  for (int l = wave; l < len; l += 8) {
    int r2 = (dir == 0) ? (l + 1) : (l - 1);   // fw: next row; bw: prev row
    float dot = 0.f;
    if (r2 >= 0 && r2 < L_ && r2 < len) {      // target masked-out => dot = 0
      const float* h = hidden + (size_t)(b * L_ + l) * (2 * D_) + dir * D_;
      const float* t = feats  + (size_t)(b * L_ + r2) * D_;
#pragma unroll
      for (int d = lane; d < D_; d += 32) dot += h[d] * t[d];
#pragma unroll
      for (int off = 16; off; off >>= 1) dot += __shfl_xor(dot, off, 32);
    }
    acc += __expf(dot) / den[dir * NROWS + b * L_ + l];
  }
  if (lane == 0) sacc[wave] = acc;
  __syncthreads();
  if (threadIdx.x == 0) {
    float s = 0.f;
#pragma unroll
    for (int w = 0; w < 8; ++w) s += sacc[w];
    partial[dir * B_ + b] = -logf(s) / (float)len * (1.0f / (float)B_);
  }
}

// ---------------------------------------------------------------------------
// Kernel 4: out[0] = fw_loss, out[1] = bw_loss.
// ---------------------------------------------------------------------------
__global__ void final_kernel(const float* __restrict__ partial,
                             float* __restrict__ out) {
  int t = threadIdx.x;
  if (t < 2) {
    float s = 0.f;
    for (int b = 0; b < B_; ++b) s += partial[t * B_ + b];
    out[t] = s;
  }
}

extern "C" void kernel_launch(void* const* d_in, const int* in_sizes, int n_in,
                              void* d_out, int out_size, void* d_ws, size_t ws_size,
                              hipStream_t stream) {
  const float*     feats  = (const float*)d_in[0];
  const float*     hidden = (const float*)d_in[1];
  const long long* lens   = (const long long*)d_in[2];

  char* ws = (char*)d_ws;
  _Float16* fh = (_Float16*)ws;                                   // 4 MB
  _Float16* hh = (_Float16*)(ws + (size_t)NROWS * D_ * 2);        // 8 MB
  float* den     = (float*)(ws + (size_t)NROWS * D_ * 2 + (size_t)NROWS * 2 * D_ * 2);
  float* partial = den + 2 * NROWS;                               // 32 floats
  float* out = (float*)d_out;

  pack_kernel<<<(NROWS * 2 * D_ + 255) / 256, 256, 0, stream>>>(feats, hidden, lens, fh, hh);

  dim3 g2(NROWS / (16 * 8), 2);   // 64 x 2 blocks, 8 row-tiles per block
  den_kernel<<<g2, 256, 0, stream>>>(fh, hh, den);

  dim3 g3(B_, 2);
  seq_kernel<<<g3, 256, 0, stream>>>(feats, hidden, lens, den, partial);

  final_kernel<<<1, 32, 0, stream>>>(partial, out);
}
